// Decoder_26173530702067
// MI455X (gfx1250) — compile-verified, matrix operand order checked by reading, not run on hardware
//
#include <hip/hip_runtime.h>
#include <hip/hip_bf16.h>
#include <math.h>

#define BSZ 32
#define T_ENC 512
#define T_DEC 512
#define N_MELS 80
#define KPAD1 96          // prenet layer-1 K padded 80 -> 96 (multiple of 32)
#define PRENET_DIM 256
#define ENC_HID 512
#define ATTN_HID 1024
#define DEC_HID 1024
#define ATTN_D 128
#define N_FILT 32
#define KSZ 31
#define PADC 15

typedef __bf16 bf16_t;
typedef bf16_t v16bf __attribute__((ext_vector_type(16)));
typedef bf16_t v8bf  __attribute__((ext_vector_type(8)));
typedef bf16_t v4bf  __attribute__((ext_vector_type(4)));
typedef float  v8f   __attribute__((ext_vector_type(8)));

// ---------------------------------------------------------------------------
// A-fragment loader: row-major fp32 activations, cvt->bf16 in-register.
// Per CDNA5 16-bit 16x32 layout a lane's fragment is two contiguous 8-elem
// runs at K-offsets hi*8 and 16+hi*8 -> 4x float4 loads, no branches.
// Requires K % 32 == 0 and 16B-aligned rows (all call sites satisfy).
// ---------------------------------------------------------------------------
__device__ __forceinline__ v16bf load_frag_a(const float* __restrict__ base,
                                             int ld, int kbase) {
  int lane = threadIdx.x & 31;
  int r    = lane & 15;
  int hi   = lane >> 4;
  const float* p = base + (size_t)r * ld + kbase + hi * 8;
  const float4 c0 = *reinterpret_cast<const float4*>(p);
  const float4 c1 = *reinterpret_cast<const float4*>(p + 4);
  const float4 c2 = *reinterpret_cast<const float4*>(p + 16);
  const float4 c3 = *reinterpret_cast<const float4*>(p + 20);
  v16bf o;
  o[0]  = (bf16_t)c0.x; o[1]  = (bf16_t)c0.y; o[2]  = (bf16_t)c0.z; o[3]  = (bf16_t)c0.w;
  o[4]  = (bf16_t)c1.x; o[5]  = (bf16_t)c1.y; o[6]  = (bf16_t)c1.z; o[7]  = (bf16_t)c1.w;
  o[8]  = (bf16_t)c2.x; o[9]  = (bf16_t)c2.y; o[10] = (bf16_t)c2.z; o[11] = (bf16_t)c2.w;
  o[12] = (bf16_t)c3.x; o[13] = (bf16_t)c3.y; o[14] = (bf16_t)c3.z; o[15] = (bf16_t)c3.w;
  return o;
}

// B-fragment loader: weights pre-converted to bf16 (row-major N x K).
// Two 16-byte runs -> 2x global_load_b128, zero converts in the hot loop.
__device__ __forceinline__ v16bf load_frag_w(const bf16_t* __restrict__ base,
                                             int ld, int kbase) {
  int lane = threadIdx.x & 31;
  int r    = lane & 15;
  int hi   = lane >> 4;
  const bf16_t* p = base + (size_t)r * ld + kbase + hi * 8;
  v8bf lo = *reinterpret_cast<const v8bf*>(p);
  v8bf hh = *reinterpret_cast<const v8bf*>(p + 16);
  v16bf o;
#pragma unroll
  for (int j = 0; j < 8; ++j) { o[j] = lo[j]; o[j + 8] = hh[j]; }
  return o;
}

// ---------------------------------------------------------------------------
// Generic WMMA GEMM:  C[M x N] = sum_i X_i[M x K_i] * W_i[N x K_i]^T + bias
// X fp32, W bf16 (pre-converted). Up to 3 (X,W) pairs (concatenated LSTM
// inputs without concat copies). One wave per 16x16 tile. act: 0 none, 1 relu.
// ---------------------------------------------------------------------------
__global__ void wmma_gemm3(
    const float* __restrict__ X0, int ldx0, const bf16_t* __restrict__ W0, int ldw0, int K0,
    const float* __restrict__ X1, int ldx1, const bf16_t* __restrict__ W1, int ldw1, int K1,
    const float* __restrict__ X2, int ldx2, const bf16_t* __restrict__ W2, int ldw2, int K2,
    const float* __restrict__ bias0, const float* __restrict__ bias1,
    float* __restrict__ C, int ldc, int N, int act) {
  int wave   = threadIdx.x >> 5;
  int tile_n = blockIdx.x * 4 + wave;
  int tile_m = blockIdx.y;
  if (tile_n * 16 >= N) return;
  int lane = threadIdx.x & 31;

  v8f acc = {};
  {
    const float*  a = X0 + (size_t)(tile_m * 16) * ldx0;
    const bf16_t* b = W0 + (size_t)(tile_n * 16) * ldw0;
    for (int k = 0; k < K0; k += 32) {
      v16bf af = load_frag_a(a, ldx0, k);
      v16bf bf = load_frag_w(b, ldw0, k);
      acc = __builtin_amdgcn_wmma_f32_16x16x32_bf16(false, af, false, bf,
                                                    (short)0, acc, false, false);
    }
  }
  if (K1 > 0) {
    const float*  a = X1 + (size_t)(tile_m * 16) * ldx1;
    const bf16_t* b = W1 + (size_t)(tile_n * 16) * ldw1;
    for (int k = 0; k < K1; k += 32) {
      v16bf af = load_frag_a(a, ldx1, k);
      v16bf bf = load_frag_w(b, ldw1, k);
      acc = __builtin_amdgcn_wmma_f32_16x16x32_bf16(false, af, false, bf,
                                                    (short)0, acc, false, false);
    }
  }
  if (K2 > 0) {
    const float*  a = X2 + (size_t)(tile_m * 16) * ldx2;
    const bf16_t* b = W2 + (size_t)(tile_n * 16) * ldw2;
    for (int k = 0; k < K2; k += 32) {
      v16bf af = load_frag_a(a, ldx2, k);
      v16bf bf = load_frag_w(b, ldw2, k);
      acc = __builtin_amdgcn_wmma_f32_16x16x32_bf16(false, af, false, bf,
                                                    (short)0, acc, false, false);
    }
  }

  int n = tile_n * 16 + (lane & 15);
  float bsum = 0.0f;
  if (bias0) bsum += bias0[n];
  if (bias1) bsum += bias1[n];
  int hi = lane >> 4;
#pragma unroll
  for (int r = 0; r < 8; ++r) {
    int m = tile_m * 16 + (hi << 3) + r;  // C/D tile layout: VGPR r -> M = r + 8*hi
    float v = acc[r] + bsum;
    if (act == 1) v = fmaxf(v, 0.0f);
    C[(size_t)m * ldc + n] = v;
  }
}

// ---------------------------------------------------------------------------
// Prep / pointwise kernels
// ---------------------------------------------------------------------------
// fp32 -> bf16 weight conversion, 4 elems per thread (n % 4 == 0).
__global__ void cvt_bf16x4(const float* __restrict__ src, bf16_t* __restrict__ dst, int n4) {
  int idx = blockIdx.x * blockDim.x + threadIdx.x;
  if (idx >= n4) return;
  float4 v = reinterpret_cast<const float4*>(src)[idx];
  v4bf o;
  o[0] = (bf16_t)v.x; o[1] = (bf16_t)v.y; o[2] = (bf16_t)v.z; o[3] = (bf16_t)v.w;
  reinterpret_cast<v4bf*>(dst)[idx] = o;
}

// Shifted decoder input, K-padded to 96 (cols 80..95 zero), fp32 (A side).
__global__ void prep_decin(const float* __restrict__ targets, float* __restrict__ dec_in) {
  int idx = blockIdx.x * blockDim.x + threadIdx.x;
  if (idx >= BSZ * T_DEC * KPAD1) return;
  int k = idx % KPAD1;
  int t = (idx / KPAD1) % T_DEC;
  int b = idx / (KPAD1 * T_DEC);
  float v = 0.0f;
  if (k < N_MELS && t > 0)
    v = targets[(size_t)b * T_DEC * N_MELS + (size_t)(t - 1) * N_MELS + k];
  dec_in[idx] = v;
}

// prenet_w1 (256 x 80) -> K-padded bf16 (256 x 96)
__global__ void prep_w1pad(const float* __restrict__ w1, bf16_t* __restrict__ w1p) {
  int idx = blockIdx.x * blockDim.x + threadIdx.x;
  if (idx >= PRENET_DIM * KPAD1) return;
  int k = idx % KPAD1;
  int n = idx / KPAD1;
  w1p[idx] = (k < N_MELS) ? (bf16_t)w1[n * N_MELS + k] : (bf16_t)0.0f;
}

// K_eff[d][c][k] = sum_f L_w[d][f] * conv_w[f][c][k]  (fold einsum into conv)
__global__ void prep_keff(const float* __restrict__ L_w, const float* __restrict__ conv_w,
                          float* __restrict__ keff) {
  int idx = blockIdx.x * blockDim.x + threadIdx.x;
  if (idx >= ATTN_D * 2 * KSZ) return;
  int k = idx % KSZ;
  int c = (idx / KSZ) & 1;
  int d = idx / (2 * KSZ);
  float s = 0.0f;
  for (int f = 0; f < N_FILT; ++f)
    s += L_w[d * N_FILT + f] * conv_w[f * 2 * KSZ + c * KSZ + k];
  keff[idx] = s;
}

__global__ void lstm_point(const float* __restrict__ gates,
                           float* __restrict__ h, float* __restrict__ c, int H) {
  int idx = blockIdx.x * blockDim.x + threadIdx.x;
  if (idx >= BSZ * H) return;
  int b = idx / H, j = idx % H;
  const float* g = gates + (size_t)b * 4 * H;
  float gi = 1.0f / (1.0f + __expf(-g[j]));
  float gf = 1.0f / (1.0f + __expf(-g[H + j]));
  float gg = tanhf(g[2 * H + j]);
  float go = 1.0f / (1.0f + __expf(-g[3 * H + j]));
  float c2 = gf * c[idx] + gi * gg;
  c[idx] = c2;
  h[idx] = go * tanhf(c2);
}

// Fused location conv + tanh energy + dot with W_w. One wave per (b, t_enc),
// each lane covers 4 of the 128 attention dims, wave-reduce at the end.
__global__ void energies_kernel(const float* __restrict__ q, const float* __restrict__ penc,
                                const float* __restrict__ keff, const float* __restrict__ Ww,
                                const float* __restrict__ aw, const float* __restrict__ aws,
                                float* __restrict__ energ) {
  int wave = threadIdx.x >> 5;
  int gid  = blockIdx.x * (blockDim.x >> 5) + wave;
  if (gid >= BSZ * T_ENC) return;
  int b = gid >> 9;
  int t = gid & (T_ENC - 1);
  int lane = threadIdx.x & 31;
  const float* awb  = aw  + b * T_ENC;
  const float* awsb = aws + b * T_ENC;
  float acc = 0.0f;
#pragma unroll
  for (int dd = 0; dd < 4; ++dd) {
    int d = lane + dd * 32;
    const float* kd = keff + d * 2 * KSZ;
    float pa = 0.0f;
    for (int k = 0; k < KSZ; ++k) {
      int tt = t - PADC + k;
      if (tt >= 0 && tt < T_ENC)
        pa += kd[k] * awb[tt] + kd[KSZ + k] * awsb[tt];
    }
    float val = q[b * ATTN_D + d] + penc[((size_t)b * T_ENC + t) * ATTN_D + d] + pa;
    acc += Ww[d] * tanhf(val);
  }
  for (int s = 16; s > 0; s >>= 1) acc += __shfl_down(acc, s, 32);
  if (lane == 0) energ[gid] = acc;
}

// Softmax over T_ENC per batch row; updates aw, aws and writes alignments.
__global__ void softmax_kernel(const float* __restrict__ energ,
                               float* __restrict__ aw, float* __restrict__ aws,
                               float* __restrict__ aligns, int t_step) {
  int b = blockIdx.x;
  int tid = threadIdx.x;  // 256 threads, 2 elems each
  __shared__ float red[256];
  const float* e = energ + b * T_ENC;
  float e0 = e[tid], e1 = e[tid + 256];
  red[tid] = fmaxf(e0, e1);
  __syncthreads();
  for (int s = 128; s > 0; s >>= 1) { if (tid < s) red[tid] = fmaxf(red[tid], red[tid + s]); __syncthreads(); }
  float m = red[0];
  __syncthreads();
  float x0 = __expf(e0 - m), x1 = __expf(e1 - m);
  red[tid] = x0 + x1;
  __syncthreads();
  for (int s = 128; s > 0; s >>= 1) { if (tid < s) red[tid] += red[tid + s]; __syncthreads(); }
  float inv = 1.0f / red[0];
  float a0 = x0 * inv, a1 = x1 * inv;
  aw[b * T_ENC + tid]        = a0;
  aw[b * T_ENC + tid + 256]  = a1;
  aws[b * T_ENC + tid]       += a0;
  aws[b * T_ENC + tid + 256] += a1;
  float* al = aligns + (size_t)b * T_DEC * T_ENC + (size_t)t_step * T_ENC;
  al[tid]       = a0;
  al[tid + 256] = a1;
}

// ctx[b,e] = sum_t aw[b,t] * enc[b,t,e]
__global__ void ctx_kernel(const float* __restrict__ aw, const float* __restrict__ enc,
                           float* __restrict__ ctx) {
  int idx = blockIdx.x * blockDim.x + threadIdx.x;
  if (idx >= BSZ * ENC_HID) return;
  int b = idx / ENC_HID, e = idx % ENC_HID;
  const float* a  = aw + b * T_ENC;
  const float* ep = enc + (size_t)b * T_ENC * ENC_HID + e;
  float s = 0.0f;
  for (int t = 0; t < T_ENC; ++t) s += a[t] * ep[(size_t)t * ENC_HID];
  ctx[idx] = s;
}

// ---------------------------------------------------------------------------
extern "C" void kernel_launch(void* const* d_in, const int* in_sizes, int n_in,
                              void* d_out, int out_size, void* d_ws, size_t ws_size,
                              hipStream_t stream) {
  const float* enc       = (const float*)d_in[0];
  const float* targets   = (const float*)d_in[1];
  const float* prenet_w1 = (const float*)d_in[2];
  const float* prenet_b1 = (const float*)d_in[3];
  const float* prenet_w2 = (const float*)d_in[4];
  const float* prenet_b2 = (const float*)d_in[5];
  const float* M_w       = (const float*)d_in[6];
  const float* Q_w       = (const float*)d_in[7];
  const float* W_w       = (const float*)d_in[8];
  const float* L_w       = (const float*)d_in[9];
  const float* conv_w    = (const float*)d_in[10];
  const float* attn_wih  = (const float*)d_in[11];
  const float* attn_whh  = (const float*)d_in[12];
  const float* attn_bih  = (const float*)d_in[13];
  const float* attn_bhh  = (const float*)d_in[14];
  const float* dec_wih   = (const float*)d_in[15];
  const float* dec_whh   = (const float*)d_in[16];
  const float* dec_bih   = (const float*)d_in[17];
  const float* dec_bhh   = (const float*)d_in[18];
  const float* proj_w    = (const float*)d_in[19];
  const float* proj_b    = (const float*)d_in[20];

  float* mel    = (float*)d_out;                                 // (B, T_DEC, 80)
  float* aligns = (float*)d_out + (size_t)BSZ * T_DEC * N_MELS;  // (B, T_DEC, 512)

  // --- workspace carve-up (fp32 region; all sizes multiples of 4 floats) ---
  float* p = (float*)d_ws;
  float* dec_in = p;      p += (size_t)BSZ * T_DEC * KPAD1;
  float* h1     = p;      p += (size_t)BSZ * T_DEC * PRENET_DIM;
  float* pre    = p;      p += (size_t)BSZ * T_DEC * PRENET_DIM;
  float* penc   = p;      p += (size_t)BSZ * T_ENC * ATTN_D;
  float* keff   = p;      p += (size_t)ATTN_D * 2 * KSZ;   // 7936, 16B-multiple
  float* state  = p;                     // zero-init region start
  float* h_a    = p;      p += BSZ * ATTN_HID;
  float* c_a    = p;      p += BSZ * ATTN_HID;
  float* h_d    = p;      p += BSZ * DEC_HID;
  float* c_d    = p;      p += BSZ * DEC_HID;
  float* aw     = p;      p += BSZ * T_ENC;
  float* aws    = p;      p += BSZ * T_ENC;
  float* ctx    = p;      p += BSZ * ENC_HID;
  size_t state_floats = (size_t)(p - state);
  float* q      = p;      p += BSZ * ATTN_D;
  float* gates_a= p;      p += BSZ * 4 * ATTN_HID;
  float* gates_d= p;      p += BSZ * 4 * DEC_HID;
  float* energ  = p;      p += BSZ * T_ENC;

  // --- bf16 weight region (element counts multiples of 8 -> 16B aligned) ---
  bf16_t* bp = (bf16_t*)p;
  bf16_t* w1p_bf   = bp;  bp += (size_t)PRENET_DIM * KPAD1;            // 24576
  bf16_t* w2_bf    = bp;  bp += (size_t)PRENET_DIM * PRENET_DIM;       // 65536
  bf16_t* Mw_bf    = bp;  bp += (size_t)ATTN_D * ENC_HID;              // 65536
  bf16_t* Qw_bf    = bp;  bp += (size_t)ATTN_D * ATTN_HID;             // 131072
  bf16_t* awih_bf  = bp;  bp += (size_t)4 * ATTN_HID * (PRENET_DIM + ENC_HID);
  bf16_t* awhh_bf  = bp;  bp += (size_t)4 * ATTN_HID * ATTN_HID;
  bf16_t* dwih_bf  = bp;  bp += (size_t)4 * DEC_HID * (ATTN_HID + ENC_HID);
  bf16_t* dwhh_bf  = bp;  bp += (size_t)4 * DEC_HID * DEC_HID;
  bf16_t* proj_bf  = bp;  bp += (size_t)N_MELS * (DEC_HID + ENC_HID);

  hipMemsetAsync(state, 0, state_floats * sizeof(float), stream);

  // --- one-time prep: shifted input, weight conversions, effective conv ---
  prep_decin<<<(BSZ * T_DEC * KPAD1 + 255) / 256, 256, 0, stream>>>(targets, dec_in);
  prep_w1pad<<<(PRENET_DIM * KPAD1 + 255) / 256, 256, 0, stream>>>(prenet_w1, w1p_bf);
  prep_keff<<<(ATTN_D * 2 * KSZ + 255) / 256, 256, 0, stream>>>(L_w, conv_w, keff);
  {
    struct { const float* s; bf16_t* d; size_t n; } cv[] = {
      { prenet_w2, w2_bf,   (size_t)PRENET_DIM * PRENET_DIM },
      { M_w,       Mw_bf,   (size_t)ATTN_D * ENC_HID },
      { Q_w,       Qw_bf,   (size_t)ATTN_D * ATTN_HID },
      { attn_wih,  awih_bf, (size_t)4 * ATTN_HID * (PRENET_DIM + ENC_HID) },
      { attn_whh,  awhh_bf, (size_t)4 * ATTN_HID * ATTN_HID },
      { dec_wih,   dwih_bf, (size_t)4 * DEC_HID * (ATTN_HID + ENC_HID) },
      { dec_whh,   dwhh_bf, (size_t)4 * DEC_HID * DEC_HID },
      { proj_w,    proj_bf, (size_t)N_MELS * (DEC_HID + ENC_HID) },
    };
    for (auto& c : cv) {
      int n4 = (int)(c.n / 4);
      cvt_bf16x4<<<(n4 + 255) / 256, 256, 0, stream>>>(c.s, c.d, n4);
    }
  }

  // prenet layer 1: (B*T x 96) @ (256 x 96)^T -> relu
  wmma_gemm3<<<dim3((PRENET_DIM / 16 + 3) / 4, (BSZ * T_DEC) / 16), 128, 0, stream>>>(
      dec_in, KPAD1, w1p_bf, KPAD1, KPAD1,
      nullptr, 0, nullptr, 0, 0, nullptr, 0, nullptr, 0, 0,
      prenet_b1, nullptr, h1, PRENET_DIM, PRENET_DIM, 1);
  // prenet layer 2
  wmma_gemm3<<<dim3((PRENET_DIM / 16 + 3) / 4, (BSZ * T_DEC) / 16), 128, 0, stream>>>(
      h1, PRENET_DIM, w2_bf, PRENET_DIM, PRENET_DIM,
      nullptr, 0, nullptr, 0, 0, nullptr, 0, nullptr, 0, 0,
      prenet_b2, nullptr, pre, PRENET_DIM, PRENET_DIM, 1);
  // processed_enc = enc @ M_w^T   (B*T_enc x 512) @ (128 x 512)^T
  wmma_gemm3<<<dim3((ATTN_D / 16 + 3) / 4, (BSZ * T_ENC) / 16), 128, 0, stream>>>(
      enc, ENC_HID, Mw_bf, ENC_HID, ENC_HID,
      nullptr, 0, nullptr, 0, 0, nullptr, 0, nullptr, 0, 0,
      nullptr, nullptr, penc, ATTN_D, ATTN_D, 0);

  for (int t = 0; t < T_DEC; ++t) {
    // attention LSTM gates: [x_t ; ctx] @ wih^T + h_a @ whh^T + biases
    wmma_gemm3<<<dim3((4 * ATTN_HID / 16 + 3) / 4, BSZ / 16), 128, 0, stream>>>(
        pre + (size_t)t * PRENET_DIM, T_DEC * PRENET_DIM,
        awih_bf, PRENET_DIM + ENC_HID, PRENET_DIM,
        ctx, ENC_HID, awih_bf + PRENET_DIM, PRENET_DIM + ENC_HID, ENC_HID,
        h_a, ATTN_HID, awhh_bf, ATTN_HID, ATTN_HID,
        attn_bih, attn_bhh, gates_a, 4 * ATTN_HID, 4 * ATTN_HID, 0);
    lstm_point<<<(BSZ * ATTN_HID + 255) / 256, 256, 0, stream>>>(gates_a, h_a, c_a, ATTN_HID);

    // q = h_a @ Q_w^T
    wmma_gemm3<<<dim3((ATTN_D / 16 + 3) / 4, BSZ / 16), 128, 0, stream>>>(
        h_a, ATTN_HID, Qw_bf, ATTN_HID, ATTN_HID,
        nullptr, 0, nullptr, 0, 0, nullptr, 0, nullptr, 0, 0,
        nullptr, nullptr, q, ATTN_D, ATTN_D, 0);

    energies_kernel<<<(BSZ * T_ENC) / 4, 128, 0, stream>>>(q, penc, keff, W_w, aw, aws, energ);
    softmax_kernel<<<BSZ, 256, 0, stream>>>(energ, aw, aws, aligns, t);
    ctx_kernel<<<(BSZ * ENC_HID + 255) / 256, 256, 0, stream>>>(aw, enc, ctx);

    // decoder LSTM gates: [h_a ; ctx] @ dec_wih^T + h_d @ dec_whh^T + biases
    wmma_gemm3<<<dim3((4 * DEC_HID / 16 + 3) / 4, BSZ / 16), 128, 0, stream>>>(
        h_a, ATTN_HID, dwih_bf, ATTN_HID + ENC_HID, ATTN_HID,
        ctx, ENC_HID, dwih_bf + ATTN_HID, ATTN_HID + ENC_HID, ENC_HID,
        h_d, DEC_HID, dwhh_bf, DEC_HID, DEC_HID,
        dec_bih, dec_bhh, gates_d, 4 * DEC_HID, 4 * DEC_HID, 0);
    lstm_point<<<(BSZ * DEC_HID + 255) / 256, 256, 0, stream>>>(gates_d, h_d, c_d, DEC_HID);

    // projection: [h_d ; ctx] @ proj_w^T + proj_b  -> mel[:, t, :]
    wmma_gemm3<<<dim3((N_MELS / 16 + 3) / 4, BSZ / 16), 128, 0, stream>>>(
        h_d, DEC_HID, proj_bf, DEC_HID + ENC_HID, DEC_HID,
        ctx, ENC_HID, proj_bf + DEC_HID, DEC_HID + ENC_HID, ENC_HID,
        nullptr, 0, nullptr, 0, 0,
        proj_b, nullptr, mel + (size_t)t * N_MELS, T_DEC * N_MELS, N_MELS, 0);
  }
}
